// VisionTransformerBlock_41592463295155
// MI455X (gfx1250) — matrix operand order repaired and checked by reading, not compiled, not run on hardware
//
#include <hip/hip_runtime.h>
#include <hip/hip_bf16.h>
#include <cstdint>

// ---------------------------------------------------------------------------
// ViT block for MI455X (gfx1250, wave32, WMMA).
// GEMMs + attention use v_wmma_f32_16x16x32_f16; tiles staged to LDS with
// gfx1250 async LDS loads (global_load_async_to_lds_b128) when available,
// double-buffered for depth-1 software pipelining.
// Shapes: B=2, N=2048, d=768, H=12, dh=64, ff=3072.
// ---------------------------------------------------------------------------

#define D_MODEL 768
#define N_HEADS 12
#define HEAD_DIM 64
#define D_FF 3072
#define SEQ 2048
#define BATCH 2
#define TROWS (BATCH * SEQ)   // 4096
#define LN_EPS 1e-5f

typedef __attribute__((ext_vector_type(16))) _Float16 v16h;
typedef __attribute__((ext_vector_type(8)))  float    v8f;
typedef __attribute__((ext_vector_type(4)))  int      v4i;

union Frag16 {                 // 8 VGPRs = 16 halfs (A or B fragment)
    v16h v;
    unsigned u[8];
};

// k-pair base inside a 16x32 f16 A-fragment for VGPR i, lane-half hi
__device__ __forceinline__ int a_kpair(int i, int hi) {
    return (i < 4 ? 2 * i : 16 + 2 * (i - 4)) + hi * 8;
}
// k-pair base inside a 32x16 f16 B-fragment for VGPR i, lane-half hi
__device__ __forceinline__ int b_kpair(int i, int hi) {
    return (i < 4 ? 2 * i : 16 + 2 * (i - 4)) + hi * 8;
}

// ---------------------------------------------------------------------------
// gfx1250 async global->LDS copy (16 bytes per active lane), with fallback.
// Builtin signature (from clang diagnostic): param0 = addrspace(1) v4i*,
// param1 = addrspace(3) v4i*, then two immediate ints (offset, cpol).
// ---------------------------------------------------------------------------
#if defined(__has_builtin)
#if __has_builtin(__builtin_amdgcn_global_load_async_to_lds_b128)
#define VIT_ASYNC_LDS 1
#endif
#endif
#ifndef VIT_ASYNC_LDS
#define VIT_ASYNC_LDS 0
#endif

__device__ __forceinline__ void vit_copy16(const _Float16* g, _Float16* l) {
#if VIT_ASYNC_LDS
    __builtin_amdgcn_global_load_async_to_lds_b128(
        (__attribute__((address_space(1))) v4i*)(void*)(g),
        (__attribute__((address_space(3))) v4i*)(void*)(l), 0, 0);
#else
    *(uint4*)l = *(const uint4*)g;
#endif
}

__device__ __forceinline__ void vit_async_wait() {
#if VIT_ASYNC_LDS
#if defined(__has_builtin) && __has_builtin(__builtin_amdgcn_s_wait_asynccnt)
    __builtin_amdgcn_s_wait_asynccnt(0);
#else
    asm volatile("s_wait_asynccnt 0" ::: "memory");
#endif
#endif
}

// ---------------------------------------------------------------------------
// 1) fp32 [K][N] -> f16 [N][K] transposing conversion (weights)
// ---------------------------------------------------------------------------
__global__ __launch_bounds__(256)
void vit_cvt_transpose(const float* __restrict__ src, _Float16* __restrict__ dst,
                       int K, int N) {
    __shared__ float tile[32][33];
    const int bn = blockIdx.x * 32;          // N block
    const int bk = blockIdx.y * 32;          // K block
    const int tx = threadIdx.x & 31;
    const int ty = threadIdx.x >> 5;         // 0..7
#pragma unroll
    for (int i = 0; i < 4; ++i) {
        int k = bk + ty + i * 8;
        tile[ty + i * 8][tx] = src[(size_t)k * N + bn + tx];
    }
    __syncthreads();
#pragma unroll
    for (int i = 0; i < 4; ++i) {
        int n = bn + ty + i * 8;
        dst[(size_t)n * K + bk + tx] = (_Float16)tile[tx][ty + i * 8];
    }
}

// ---------------------------------------------------------------------------
// 2) LayerNorm over 768 features, fp32 in -> f16 out. One block per row.
// ---------------------------------------------------------------------------
__global__ __launch_bounds__(256)
void vit_ln768(const float* __restrict__ x, const float* __restrict__ g,
               const float* __restrict__ b, _Float16* __restrict__ out) {
    __shared__ float red[256];
    const int row = blockIdx.x;
    const int t   = threadIdx.x;
    const float* xr = x + (size_t)row * D_MODEL;

    float v[3];
    float s = 0.f;
#pragma unroll
    for (int i = 0; i < 3; ++i) { v[i] = xr[t + 256 * i]; s += v[i]; }
    red[t] = s;
    __syncthreads();
    for (int off = 128; off > 0; off >>= 1) {
        if (t < off) red[t] += red[t + off];
        __syncthreads();
    }
    const float mu = red[0] * (1.f / D_MODEL);
    __syncthreads();

    s = 0.f;
#pragma unroll
    for (int i = 0; i < 3; ++i) { float d = v[i] - mu; s += d * d; }
    red[t] = s;
    __syncthreads();
    for (int off = 128; off > 0; off >>= 1) {
        if (t < off) red[t] += red[t + off];
        __syncthreads();
    }
    const float rstd = rsqrtf(red[0] * (1.f / D_MODEL) + LN_EPS);

#pragma unroll
    for (int i = 0; i < 3; ++i) {
        int c = t + 256 * i;
        out[(size_t)row * D_MODEL + c] =
            (_Float16)((v[i] - mu) * rstd * g[c] + b[c]);
    }
}

// ---------------------------------------------------------------------------
// 3) WMMA GEMM: C[M,N] = A[M,K] (f16, row major) * W^T[N,K] (f16) + epilogue.
//    64x64 tile / block, 4 waves, BK=32, double-buffered async LDS staging.
//    mode 0: out16 <- acc+bias scattered to [B,H,N,64]       (Q,K)
//    mode 4: out16 <- acc+bias scattered to [B,H,64,N]       (V, transposed)
//    mode 1: out16 <- gelu(acc+bias) row-major               (MLP fc1)
//    mode 2: out32 <- res + (acc+bias)*ls[col] row-major     (proj / fc2)
// ---------------------------------------------------------------------------
__global__ __launch_bounds__(128)
void vit_gemm_wmma(const _Float16* __restrict__ A,
                   const _Float16* __restrict__ WT,   // [N][K]
                   const float* __restrict__ bias,
                   int N, int K, int mode,
                   _Float16* __restrict__ out16,
                   float* __restrict__ out32,
                   const float* __restrict__ res,
                   const float* __restrict__ ls) {
    __shared__ __align__(16) _Float16 As[2][64][32];   // [m][k]
    __shared__ __align__(16) _Float16 Bs[2][64][32];   // [n][k]

    const int tid  = threadIdx.x;
    const int w    = tid >> 5;
    const int lane = tid & 31;
    const int l15  = lane & 15;
    const int hi   = lane >> 4;
    const int n0   = blockIdx.x * 64;
    const int m0   = blockIdx.y * 64;

    // stage one 64x32 tile pair: 2x256 16B chunks, 2+2 per thread
    auto stage = [&](int buf, int k0) {
        const _Float16* Ab = A  + (size_t)m0 * K + k0;
        const _Float16* Bb = WT + (size_t)n0 * K + k0;
#pragma unroll
        for (int i = 0; i < 2; ++i) {
            int e = tid + i * 128;
            int r = e >> 2;
            int c = (e & 3) * 8;
            vit_copy16(Ab + (size_t)r * K + c, &As[buf][r][c]);
        }
#pragma unroll
        for (int i = 0; i < 2; ++i) {
            int e = tid + i * 128;
            int r = e >> 2;
            int c = (e & 3) * 8;
            vit_copy16(Bb + (size_t)r * K + c, &Bs[buf][r][c]);
        }
    };

    v8f acc[4] = {v8f{}, v8f{}, v8f{}, v8f{}};
    const int nk = K >> 5;
    const int arow = w * 16 + l15;

    stage(0, 0);
    vit_async_wait();
    __syncthreads();

    for (int kt = 0; kt < nk; ++kt) {
        const int cur = kt & 1;
        if (kt + 1 < nk) stage(cur ^ 1, (kt + 1) * 32);   // prefetch next tile

        Frag16 af, bf;
#pragma unroll
        for (int i = 0; i < 8; ++i)
            af.u[i] = *(const unsigned*)&As[cur][arow][a_kpair(i, hi)];
#pragma unroll
        for (int j = 0; j < 4; ++j) {
            const int bn = j * 16 + l15;
#pragma unroll
            for (int i = 0; i < 8; ++i)
                bf.u[i] = *(const unsigned*)&Bs[cur][bn][b_kpair(i, hi)];
            acc[j] = __builtin_amdgcn_wmma_f32_16x16x32_f16(
                false, af.v, false, bf.v, (short)0, acc[j], false, false);
        }
        vit_async_wait();
        __syncthreads();
    }

    // epilogue (C layout: VGPR r, lanes 0-15 -> M=r, lanes 16-31 -> M=8+r)
#pragma unroll
    for (int j = 0; j < 4; ++j) {
        const int col = n0 + j * 16 + l15;
        const float bv = bias[col];
#pragma unroll
        for (int r = 0; r < 8; ++r) {
            const int row = m0 + w * 16 + hi * 8 + r;
            float val = acc[j][r] + bv;
            if (mode == 0) {
                int bidx = row >> 11, nn = row & (SEQ - 1);
                int head = col >> 6,  wi = col & 63;
                out16[(((size_t)bidx * N_HEADS + head) * SEQ + nn) * HEAD_DIM + wi] =
                    (_Float16)val;
            } else if (mode == 4) {
                int bidx = row >> 11, nn = row & (SEQ - 1);
                int head = col >> 6,  wi = col & 63;
                out16[(((size_t)bidx * N_HEADS + head) * HEAD_DIM + wi) * SEQ + nn] =
                    (_Float16)val;
            } else if (mode == 1) {
                val = 0.5f * val * (1.f + erff(val * 0.70710678118654752f));
                out16[(size_t)row * N + col] = (_Float16)val;
            } else {
                out32[(size_t)row * N + col] =
                    res[(size_t)row * N + col] + val * ls[col];
            }
        }
    }
}

// ---------------------------------------------------------------------------
// 4) Per-head LayerNorm over 64 features (f16 in-place). 1 wave per row.
// ---------------------------------------------------------------------------
__global__ __launch_bounds__(256)
void vit_headln(_Float16* __restrict__ data, const float* __restrict__ g,
                const float* __restrict__ b, int nrows) {
    const int wid  = threadIdx.x >> 5;
    const int lane = threadIdx.x & 31;
    const int row  = blockIdx.x * 8 + wid;
    if (row >= nrows) return;
    _Float16* p = data + (size_t)row * HEAD_DIM;

    float v0 = (float)p[lane * 2], v1 = (float)p[lane * 2 + 1];
    float s = v0 + v1;
#pragma unroll
    for (int off = 16; off > 0; off >>= 1) s += __shfl_xor(s, off, 32);
    const float mu = s * (1.f / HEAD_DIM);
    float d0 = v0 - mu, d1 = v1 - mu;
    float q = d0 * d0 + d1 * d1;
#pragma unroll
    for (int off = 16; off > 0; off >>= 1) q += __shfl_xor(q, off, 32);
    const float rstd = rsqrtf(q * (1.f / HEAD_DIM) + LN_EPS);

    p[lane * 2]     = (_Float16)(d0 * rstd * g[lane * 2]     + b[lane * 2]);
    p[lane * 2 + 1] = (_Float16)(d1 * rstd * g[lane * 2 + 1] + b[lane * 2 + 1]);
}

// ---------------------------------------------------------------------------
// 5) Flash attention with ALiBi-2D bias. Block = 4 waves, 64 query rows.
//    Q: [B,H,N,64]; K: [B,H,N,64]; V pre-transposed: [B,H,64,N].
//    K/V tiles double-buffered through async LDS loads.
// ---------------------------------------------------------------------------
__global__ __launch_bounds__(128)
void vit_attention(const _Float16* __restrict__ Qm,
                   const _Float16* __restrict__ Km,
                   const _Float16* __restrict__ VTm,
                   const float* __restrict__ positions,
                   const float* __restrict__ patch_sizes,
                   const float* __restrict__ slopes,
                   const int* __restrict__ nreg_p,
                   _Float16* __restrict__ Oout) {
    __shared__ __align__(16) _Float16 Qs[64][64];       // [qrow][feat]
    __shared__ __align__(16) _Float16 Ks[2][64][64];    // [key][feat]
    __shared__ __align__(16) _Float16 Vs[2][64][64];    // [feat][key]
    __shared__ __align__(16) _Float16 Ps[4][16][64];    // per-wave P staging
    __shared__ float posqx[64], posqy[64];
    __shared__ float poskx[2][64], posky[2][64];
    __shared__ float keepq[64], keepk[2][64];

    const int tid  = threadIdx.x;
    const int w    = tid >> 5;
    const int lane = tid & 31;
    const int l15  = lane & 15;
    const int hi   = lane >> 4;

    const int bid = blockIdx.x;
    const int qt  = bid & 31;                 // SEQ/64 query tiles
    const int h   = (bid >> 5) % N_HEADS;
    const int b   = bid / (32 * N_HEADS);
    const int q0  = qt * 64;
    const size_t base  = (size_t)(b * N_HEADS + h) * SEQ * HEAD_DIM;  // Q,K
    const size_t vbase = (size_t)(b * N_HEADS + h) * HEAD_DIM * SEQ;  // V^T

    float ps = patch_sizes[b];
    ps = ps < 1e-6f ? 1e-6f : ps;
    const float inv_ps = 1.f / ps;
    const float slope  = slopes[h];
    const int nreg     = nreg_p[0];
    const float scale  = 0.125f;              // 1/sqrt(64)

    // stage K tile (contiguous) + V^T tile (rows strided by SEQ) + key meta
    auto stage_kv = [&](int buf, int k0g) {
        const _Float16* ksrc = Km + base + (size_t)k0g * HEAD_DIM;
#pragma unroll
        for (int i = 0; i < 4; ++i) {
            int e = tid + i * 128;
            vit_copy16(ksrc + e * 8, &Ks[buf][0][0] + e * 8);
        }
#pragma unroll
        for (int i = 0; i < 4; ++i) {
            int e = tid + i * 128;
            int f = e >> 3, c = (e & 7) * 8;
            vit_copy16(VTm + vbase + (size_t)f * SEQ + k0g + c, &Vs[buf][f][c]);
        }
        if (tid < 64) {
            int idx = k0g + tid;
            poskx[buf][tid] = positions[((size_t)b * SEQ + idx) * 2 + 0];
            posky[buf][tid] = positions[((size_t)b * SEQ + idx) * 2 + 1];
            bool special = (idx == 0) | ((idx >= 1) & (idx < 1 + nreg));
            keepk[buf][tid] = special ? 0.f : 1.f;
        }
    };

    // stage Q tile (fully contiguous) + query meta
    {
        const _Float16* qsrc = Qm + base + (size_t)q0 * HEAD_DIM;
#pragma unroll
        for (int i = 0; i < 4; ++i) {
            int e = tid + i * 128;
            vit_copy16(qsrc + e * 8, &Qs[0][0] + e * 8);
        }
        if (tid < 64) {
            int idx = q0 + tid;
            posqx[tid] = positions[((size_t)b * SEQ + idx) * 2 + 0];
            posqy[tid] = positions[((size_t)b * SEQ + idx) * 2 + 1];
            bool special = (idx == 0) | ((idx >= 1) & (idx < 1 + nreg));
            keepq[tid] = special ? 0.f : 1.f;
        }
    }
    stage_kv(0, 0);
    vit_async_wait();
    __syncthreads();

    // hoist Q A-fragments (tile-invariant) into registers
    const int arow = w * 16 + l15;
    Frag16 aq[2];
#pragma unroll
    for (int ks = 0; ks < 2; ++ks)
#pragma unroll
        for (int i = 0; i < 8; ++i)
            aq[ks].u[i] = *(const unsigned*)&Qs[arow][a_kpair(i, hi) + ks * 32];

    v8f acc_o[4] = {v8f{}, v8f{}, v8f{}, v8f{}};
    float m_run[8], l_run[8];
#pragma unroll
    for (int r = 0; r < 8; ++r) { m_run[r] = -3e38f; l_run[r] = 0.f; }

    for (int kt = 0; kt < SEQ / 64; ++kt) {
        const int cur = kt & 1;
        if (kt + 1 < SEQ / 64) stage_kv(cur ^ 1, (kt + 1) * 64);

        // S = Q * K^T  (16 rows x 64 keys per wave)
        v8f s[4] = {v8f{}, v8f{}, v8f{}, v8f{}};
        Frag16 bf;
#pragma unroll
        for (int ks = 0; ks < 2; ++ks) {
#pragma unroll
            for (int j = 0; j < 4; ++j) {
                const int key = j * 16 + l15;
#pragma unroll
                for (int i = 0; i < 8; ++i)
                    bf.u[i] = *(const unsigned*)&Ks[cur][key][b_kpair(i, hi) + ks * 32];
                s[j] = __builtin_amdgcn_wmma_f32_16x16x32_f16(
                    false, aq[ks].v, false, bf.v, (short)0, s[j], false, false);
            }
        }

        // scale + ALiBi bias + mask; per-row tile max
        float tmax[8];
#pragma unroll
        for (int r = 0; r < 8; ++r) tmax[r] = -3e38f;
#pragma unroll
        for (int j = 0; j < 4; ++j) {
            const int kj = j * 16 + l15;
            const float kx = poskx[cur][kj], ky = posky[cur][kj], kk = keepk[cur][kj];
#pragma unroll
            for (int r = 0; r < 8; ++r) {
                const int qi = w * 16 + hi * 8 + r;
                float dx = posqx[qi] - kx, dy = posqy[qi] - ky;
                float biasv = slope * sqrtf(dx * dx + dy * dy) * inv_ps *
                              keepq[qi] * kk;
                float v = s[j][r] * scale + biasv;
                s[j][r] = v;
                tmax[r] = fmaxf(tmax[r], v);
            }
        }
#pragma unroll
        for (int r = 0; r < 8; ++r)
#pragma unroll
            for (int off = 1; off < 16; off <<= 1)
                tmax[r] = fmaxf(tmax[r], __shfl_xor(tmax[r], off, 16));

        // online softmax update
        float rsum[8];
#pragma unroll
        for (int r = 0; r < 8; ++r) {
            float newm = fmaxf(m_run[r], tmax[r]);
            float sc   = expf(m_run[r] - newm);
            rsum[r] = 0.f;
#pragma unroll
            for (int j = 0; j < 4; ++j) {
                float p = expf(s[j][r] - newm);
                s[j][r] = p;
                rsum[r] += p;
            }
#pragma unroll
            for (int j = 0; j < 4; ++j) acc_o[j][r] *= sc;
            l_run[r] *= sc;
            m_run[r] = newm;
        }
#pragma unroll
        for (int r = 0; r < 8; ++r) {
#pragma unroll
            for (int off = 1; off < 16; off <<= 1)
                rsum[r] += __shfl_xor(rsum[r], off, 16);
            l_run[r] += rsum[r];
        }

        // C-layout -> A-layout relayout of P through per-wave LDS
#pragma unroll
        for (int j = 0; j < 4; ++j)
#pragma unroll
            for (int r = 0; r < 8; ++r)
                Ps[w][hi * 8 + r][j * 16 + l15] = (_Float16)s[j][r];
        // same-wave DS ops are in-order: safe to reload immediately

        // O += P * V
        Frag16 af;
#pragma unroll
        for (int ks = 0; ks < 2; ++ks) {
#pragma unroll
            for (int i = 0; i < 8; ++i)
                af.u[i] = *(const unsigned*)&Ps[w][l15][a_kpair(i, hi) + ks * 32];
#pragma unroll
            for (int j = 0; j < 4; ++j) {
                const int nf = j * 16 + l15;
#pragma unroll
                for (int i = 0; i < 8; ++i)
                    bf.u[i] = *(const unsigned*)&Vs[cur][nf][b_kpair(i, hi) + ks * 32];
                acc_o[j] = __builtin_amdgcn_wmma_f32_16x16x32_f16(
                    false, af.v, false, bf.v, (short)0, acc_o[j], false, false);
            }
        }
        vit_async_wait();
        __syncthreads();
    }

    // normalize and write attn_out as [B, N, 768] f16
#pragma unroll
    for (int j = 0; j < 4; ++j) {
        const int col = h * HEAD_DIM + j * 16 + l15;
#pragma unroll
        for (int r = 0; r < 8; ++r) {
            const int qi = q0 + w * 16 + hi * 8 + r;
            float ov = acc_o[j][r] / l_run[r];
            Oout[((size_t)b * SEQ + qi) * D_MODEL + col] = (_Float16)ov;
        }
    }
}

// ---------------------------------------------------------------------------
// Host-side orchestration
// ---------------------------------------------------------------------------
extern "C" void kernel_launch(void* const* d_in, const int* in_sizes, int n_in,
                              void* d_out, int out_size, void* d_ws, size_t ws_size,
                              hipStream_t stream) {
    const float* x         = (const float*)d_in[0];
    const float* positions = (const float*)d_in[1];
    const float* patch     = (const float*)d_in[2];
    const int*   nreg      = (const int*)d_in[3];
    const float* slopes    = (const float*)d_in[4];
    const float* n1g = (const float*)d_in[5];
    const float* n1b = (const float*)d_in[6];
    const float* wq  = (const float*)d_in[7];
    const float* bq  = (const float*)d_in[8];
    const float* wk  = (const float*)d_in[9];
    const float* bk  = (const float*)d_in[10];
    const float* wv  = (const float*)d_in[11];
    const float* bv  = (const float*)d_in[12];
    const float* wo  = (const float*)d_in[13];
    const float* bo  = (const float*)d_in[14];
    const float* qng = (const float*)d_in[15];
    const float* qnb = (const float*)d_in[16];
    const float* kng = (const float*)d_in[17];
    const float* knb = (const float*)d_in[18];
    const float* n2g = (const float*)d_in[19];
    const float* n2b = (const float*)d_in[20];
    const float* w1  = (const float*)d_in[21];
    const float* b1  = (const float*)d_in[22];
    const float* w2  = (const float*)d_in[23];
    const float* b2  = (const float*)d_in[24];
    const float* lsa = (const float*)d_in[25];
    const float* lsm = (const float*)d_in[26];

    char* ws = (char*)d_ws;
    size_t off = 0;
    auto alloc = [&](size_t bytes) -> void* {
        void* p = ws + off;
        off += (bytes + 255) & ~(size_t)255;
        return p;
    };

    const size_t dd  = (size_t)D_MODEL * D_MODEL;
    const size_t dff = (size_t)D_MODEL * D_FF;
    _Float16* wqT = (_Float16*)alloc(dd * 2);     // [768][768]
    _Float16* wkT = (_Float16*)alloc(dd * 2);
    _Float16* wvT = (_Float16*)alloc(dd * 2);
    _Float16* woT = (_Float16*)alloc(dd * 2);
    _Float16* w1T = (_Float16*)alloc(dff * 2);    // [3072][768]
    _Float16* w2T = (_Float16*)alloc(dff * 2);    // [768][3072]
    _Float16* xn1  = (_Float16*)alloc((size_t)TROWS * D_MODEL * 2);
    _Float16* q16  = (_Float16*)alloc((size_t)TROWS * D_MODEL * 2);  // [B,H,N,64]
    _Float16* k16  = (_Float16*)alloc((size_t)TROWS * D_MODEL * 2);  // [B,H,N,64]
    _Float16* vT16 = (_Float16*)alloc((size_t)TROWS * D_MODEL * 2);  // [B,H,64,N]
    _Float16* ao16 = (_Float16*)alloc((size_t)TROWS * D_MODEL * 2);
    float*    x1   = (float*)   alloc((size_t)TROWS * D_MODEL * 4);
    _Float16* xn2  = (_Float16*)alloc((size_t)TROWS * D_MODEL * 2);
    _Float16* h16  = (_Float16*)alloc((size_t)TROWS * D_FF * 2);

    // 1) weights fp32 [K][N] -> f16 [N][K]
    auto cvtT = [&](const float* s, _Float16* d, int K, int N) {
        dim3 g(N / 32, K / 32);
        vit_cvt_transpose<<<g, 256, 0, stream>>>(s, d, K, N);
    };
    cvtT(wq, wqT, D_MODEL, D_MODEL);
    cvtT(wk, wkT, D_MODEL, D_MODEL);
    cvtT(wv, wvT, D_MODEL, D_MODEL);
    cvtT(wo, woT, D_MODEL, D_MODEL);
    cvtT(w1, w1T, D_MODEL, D_FF);
    cvtT(w2, w2T, D_FF, D_MODEL);

    // 2) LN1
    vit_ln768<<<TROWS, 256, 0, stream>>>(x, n1g, n1b, xn1);

    // 3) QKV projections
    dim3 gQKV(D_MODEL / 64, TROWS / 64);
    vit_gemm_wmma<<<gQKV, 128, 0, stream>>>(xn1, wqT, bq, D_MODEL, D_MODEL,
                                            0, q16, nullptr, nullptr, nullptr);
    vit_gemm_wmma<<<gQKV, 128, 0, stream>>>(xn1, wkT, bk, D_MODEL, D_MODEL,
                                            0, k16, nullptr, nullptr, nullptr);
    vit_gemm_wmma<<<gQKV, 128, 0, stream>>>(xn1, wvT, bv, D_MODEL, D_MODEL,
                                            4, vT16, nullptr, nullptr, nullptr);

    // 4) per-head LN on q and k
    const int hrows = BATCH * N_HEADS * SEQ;   // 49152
    vit_headln<<<hrows / 8, 256, 0, stream>>>(q16, qng, qnb, hrows);
    vit_headln<<<hrows / 8, 256, 0, stream>>>(k16, kng, knb, hrows);

    // 5) attention
    vit_attention<<<BATCH * N_HEADS * (SEQ / 64), 128, 0, stream>>>(
        q16, k16, vT16, positions, patch, slopes, nreg, ao16);

    // 6) output projection + layer scale + residual (fp32)
    vit_gemm_wmma<<<gQKV, 128, 0, stream>>>(ao16, woT, bo, D_MODEL, D_MODEL,
                                            2, nullptr, x1, x, lsa);

    // 7) LN2
    vit_ln768<<<TROWS, 256, 0, stream>>>(x1, n2g, n2b, xn2);

    // 8) MLP fc1 + GELU
    dim3 gFF(D_FF / 64, TROWS / 64);
    vit_gemm_wmma<<<gFF, 128, 0, stream>>>(xn2, w1T, b1, D_FF, D_MODEL,
                                           1, h16, nullptr, nullptr, nullptr);

    // 9) MLP fc2 + layer scale + residual -> final output (fp32)
    vit_gemm_wmma<<<gQKV, 128, 0, stream>>>(h16, w2T, b2, D_MODEL, D_FF,
                                            2, nullptr, (float*)d_out, x1, lsm);
}
